// AttentionHead_43241730736911
// MI455X (gfx1250) — matrix operand order repaired and checked by reading, not compiled, not run on hardware
//
#include <hip/hip_runtime.h>
#include <hip/hip_bf16.h>

typedef __attribute__((ext_vector_type(16))) __bf16 v16bf;
typedef __attribute__((ext_vector_type(8)))  __bf16 v8bf;
typedef __attribute__((ext_vector_type(8)))  float  v8f;
typedef __attribute__((ext_vector_type(4)))  float  v4f;

static __device__ __forceinline__ v16bf combine16(v8bf lo, v8bf hi) {
    v16bf r;
#pragma unroll
    for (int i = 0; i < 8; ++i) { r[i] = lo[i]; r[i + 8] = hi[i]; }
    return r;
}

// ---------------------------------------------------------------------------
// Kernel 1: convert Wq/Wk/Wv fp32 [C=1024, H=128] -> bf16 transposed [H, C]
// so WMMA B-fragments are contiguous along K(=C).
// ---------------------------------------------------------------------------
__global__ void wprep_kernel(const float* __restrict__ Wq,
                             const float* __restrict__ Wk,
                             const float* __restrict__ Wv,
                             __bf16* __restrict__ Wt) {
    int idx = blockIdx.x * 256 + threadIdx.x;        // 0 .. 3*131072-1
    int mat = idx >> 17;                             // 131072 = 1024*128
    int rem = idx & 131071;                          // rem = c*128 + h
    int h = rem & 127;
    int c = rem >> 7;
    const float* W = (mat == 0) ? Wq : (mat == 1) ? Wk : Wv;
    Wt[(size_t)mat * 131072 + (size_t)h * 1024 + c] = (__bf16)W[rem];
}

// ---------------------------------------------------------------------------
// Kernel 2: fused QKV projection.  X[16384,1024] @ [Wq|Wk|Wv] (N=384 total).
// grid = 256 blocks; block = 256 (8 waves).  Wave = 16(M) x 192(N) strip
// = 12 WMMA tiles; tile t -> matrix (t>>2) at compile time after unroll, so
// the Q/K/V store selection constant-folds (no runtime branches).
// Each A fragment feeds 12 WMMAs; X is read once per block.
// Q, K stored row-major bf16 [b*T+t][h]; V stored transposed bf16 [b][h][t].
// ---------------------------------------------------------------------------
__global__ void __launch_bounds__(256)
qkv_proj_kernel(const float* __restrict__ X, const __bf16* __restrict__ Wt,
                __bf16* __restrict__ Qb, __bf16* __restrict__ Kb,
                __bf16* __restrict__ Vt) {
    const int lane  = threadIdx.x & 31;
    const int wave  = threadIdx.x >> 5;
    const int mBase = blockIdx.x * 64 + (wave >> 1) * 16;
    const int nOff  = (wave & 1) * 16;   // interleaved 16-col half within each 32-col group
    const int l16   = lane & 15;
    const int half  = lane >> 4;
    const int koffA = half * 8;    // A-layout K offset (8/8 interleave)
    const int koffB = half * 16;   // contiguous K offset (unused; B also uses koffA)

    const float* xrow = X + (size_t)(mBase + l16) * 1024;

    v8f acc[12] = {};
    for (int kk = 0; kk < 1024; kk += 32) {
        // A fragment: convert fp32 -> bf16 in-flight, ISA 16-bit A layout
        v4f a0 = *(const v4f*)(xrow + kk + koffA);
        v4f a1 = *(const v4f*)(xrow + kk + koffA + 4);
        v4f a2 = *(const v4f*)(xrow + kk + 16 + koffA);
        v4f a3 = *(const v4f*)(xrow + kk + 16 + koffA + 4);
        v16bf a;
#pragma unroll
        for (int i = 0; i < 4; ++i) {
            a[i]      = (__bf16)a0[i];
            a[4 + i]  = (__bf16)a1[i];
            a[8 + i]  = (__bf16)a2[i];
            a[12 + i] = (__bf16)a3[i];
        }
        // 12 tiles in 3 groups of 4: batch the B loads, then the WMMA chain
#pragma unroll
        for (int g = 0; g < 3; ++g) {
            v16bf bfrag[4];
#pragma unroll
            for (int j = 0; j < 4; ++j) {
                const int t = g * 4 + j;
                const int mat = t >> 2;                       // 0..2
                const int col = (t & 3) * 32 + nOff + l16;    // 0..127
                const __bf16* wp = Wt + (size_t)mat * 131072 +
                                   (size_t)col * 1024 + kk + koffA;
                bfrag[j] = combine16(*(const v8bf*)wp, *(const v8bf*)(wp + 16));
            }
#pragma unroll
            for (int j = 0; j < 4; ++j) {
                acc[g * 4 + j] = __builtin_amdgcn_wmma_f32_16x16x32_bf16(
                    false, a, false, bfrag[j], (short)0, acc[g * 4 + j], false, false);
            }
        }
    }
    // Store: C/D layout = lane l holds rows r+8*half, col l16.
    // mat = t>>2 is a compile-time constant after unrolling -> no branches.
#pragma unroll
    for (int t = 0; t < 12; ++t) {
        const int col = (t & 3) * 32 + nOff + l16;
#pragma unroll
        for (int r = 0; r < 8; ++r) {
            int m = mBase + r + half * 8;
            __bf16 val = (__bf16)acc[t][r];
            if (t < 4) {
                Qb[(size_t)m * 128 + col] = val;
            } else if (t < 8) {
                Kb[(size_t)m * 128 + col] = val;
            } else {
                int b = m >> 11, tt = m & 2047;
                Vt[((size_t)b * 128 + col) * 2048 + tt] = val;
            }
        }
    }
}

// ---------------------------------------------------------------------------
// Kernel 3: flash attention.  One wave owns 16 query rows x all keys.
// grid = B * (T/128) = 128 blocks; block = 256 (8 waves, independent).
// S chunk = 16q x 32k (two WMMA D-tiles), online softmax, P bounced through
// wave-private LDS to re-layout D->A, then P@V against transposed V.
// ---------------------------------------------------------------------------
__global__ void __launch_bounds__(256)
flash_attn_kernel(const __bf16* __restrict__ Qb, const __bf16* __restrict__ Kb,
                  const __bf16* __restrict__ Vt, float* __restrict__ Out) {
    __shared__ __align__(16) __bf16 pbuf[8][16][40];  // padded rows: 16B aligned

    const int lane  = threadIdx.x & 31;
    const int wave  = threadIdx.x >> 5;
    const int b     = blockIdx.x >> 4;
    const int qbase = (blockIdx.x & 15) * 128 + wave * 16;
    const int l16   = lane & 15;
    const int half  = lane >> 4;
    const int koffA = half * 8;
    const int koffB = half * 16;
    const float scale = 0.03125f;  // 1024^-0.5 (embed-size scaling per reference)

    // Resident Q A-fragments: 16 rows x 128 (4 k-steps of 32)
    const __bf16* qrow = Qb + (size_t)(b * 2048 + qbase + l16) * 128;
    v16bf qa[4];
#pragma unroll
    for (int kk = 0; kk < 4; ++kk) {
        qa[kk] = combine16(*(const v8bf*)(qrow + kk * 32 + koffB),
                           *(const v8bf*)(qrow + kk * 32 + koffB + 8));
    }

    float rowmax[8], rowsum[8];
#pragma unroll
    for (int r = 0; r < 8; ++r) { rowmax[r] = -__builtin_inff(); rowsum[r] = 0.f; }
    v8f oacc[8] = {};  // 16 rows x 128 cols fp32

    const __bf16* kmat = Kb + (size_t)b * 2048 * 128;
    const __bf16* vmat = Vt + (size_t)b * 128 * 2048;

    const int kvEnd = qbase + 16;
    for (int kv = 0; kv < kvEnd; kv += 32) {
        // Prefetch next chunk's K rows (global_prefetch_b8) while we compute
        if (kv + 32 < kvEnd) {
            __builtin_prefetch(kmat + (size_t)(kv + 32 + l16) * 128, 0, 3);
            __builtin_prefetch(kmat + (size_t)(kv + 48 + l16) * 128, 0, 3);
        }
        // S = Q @ K^T for 32 keys: two 16x16 tiles; batch loads then WMMAs
        v8f s0 = {}, s1 = {};
        {
            v16bf bk[4];
#pragma unroll
            for (int kk = 0; kk < 4; ++kk) {
                const __bf16* kp = kmat + (size_t)(kv + l16) * 128 + kk * 32 + koffB;
                bk[kk] = combine16(*(const v8bf*)kp, *(const v8bf*)(kp + 8));
            }
#pragma unroll
            for (int kk = 0; kk < 4; ++kk)
                s0 = __builtin_amdgcn_wmma_f32_16x16x32_bf16(
                    false, qa[kk], false, bk[kk], (short)0, s0, false, false);
#pragma unroll
            for (int kk = 0; kk < 4; ++kk) {
                const __bf16* kp = kmat + (size_t)(kv + 16 + l16) * 128 + kk * 32 + koffB;
                bk[kk] = combine16(*(const v8bf*)kp, *(const v8bf*)(kp + 8));
            }
#pragma unroll
            for (int kk = 0; kk < 4; ++kk)
                s1 = __builtin_amdgcn_wmma_f32_16x16x32_bf16(
                    false, qa[kk], false, bk[kk], (short)0, s1, false, false);
        }
        // Causal mask + scale + online softmax (per-row, 16-lane reductions)
#pragma unroll
        for (int r = 0; r < 8; ++r) {
            int qrowi = qbase + r + half * 8;
            int key0 = kv + l16;
            int key1 = key0 + 16;
            float v0 = (key0 <= qrowi) ? s0[r] * scale : -__builtin_inff();
            float v1 = (key1 <= qrowi) ? s1[r] * scale : -__builtin_inff();
            float cm = fmaxf(v0, v1);
#pragma unroll
            for (int m = 8; m >= 1; m >>= 1) cm = fmaxf(cm, __shfl_xor(cm, m));
            float mnew  = fmaxf(rowmax[r], cm);
            float alpha = __expf(rowmax[r] - mnew);
            rowmax[r] = mnew;
            float p0 = __expf(v0 - mnew);
            float p1 = __expf(v1 - mnew);
            float ps = p0 + p1;
#pragma unroll
            for (int m = 8; m >= 1; m >>= 1) ps += __shfl_xor(ps, m);
            rowsum[r] = rowsum[r] * alpha + ps;
#pragma unroll
            for (int t = 0; t < 8; ++t) oacc[t][r] *= alpha;
            // stash P (D layout) into LDS for A-layout reload
            pbuf[wave][r + half * 8][l16]      = (__bf16)p0;
            pbuf[wave][r + half * 8][l16 + 16] = (__bf16)p1;
        }
        asm volatile("s_wait_dscnt 0" ::: "memory");  // wave-private LDS store->load
        v16bf pa = combine16(*(const v8bf*)&pbuf[wave][l16][koffB],
                             *(const v8bf*)&pbuf[wave][l16][koffB + 8]);
        // O += P @ V   (V transposed -> contiguous B fragments)
#pragma unroll
        for (int g = 0; g < 2; ++g) {
            v16bf vb[4];
#pragma unroll
            for (int j = 0; j < 4; ++j) {
                const int t = g * 4 + j;
                const __bf16* vp = vmat + (size_t)(t * 16 + l16) * 2048 + kv + koffB;
                vb[j] = combine16(*(const v8bf*)vp, *(const v8bf*)(vp + 8));
            }
#pragma unroll
            for (int j = 0; j < 4; ++j) {
                oacc[g * 4 + j] = __builtin_amdgcn_wmma_f32_16x16x32_bf16(
                    false, pa, false, vb[j], (short)0, oacc[g * 4 + j], false, false);
            }
        }
    }
    // Epilogue: divide by row sums, store fp32
#pragma unroll
    for (int r = 0; r < 8; ++r) {
        float inv = 1.f / rowsum[r];
        size_t rowoff = (size_t)(b * 2048 + qbase + r + half * 8) * 128;
#pragma unroll
        for (int t = 0; t < 8; ++t) {
            Out[rowoff + t * 16 + l16] = oacc[t][r] * inv;
        }
    }
}

// ---------------------------------------------------------------------------
extern "C" void kernel_launch(void* const* d_in, const int* in_sizes, int n_in,
                              void* d_out, int out_size, void* d_ws, size_t ws_size,
                              hipStream_t stream) {
    const float* X  = (const float*)d_in[0];  // [8,2048,1024]
    const float* Wq = (const float*)d_in[1];  // [1024,128]
    const float* Wk = (const float*)d_in[2];
    const float* Wv = (const float*)d_in[3];

    // workspace layout (bytes):
    //   Wt  : 3 * 128*1024 bf16      =   786432
    //   Qb  : 8*2048*128  bf16       =  4194304
    //   Kb  : 8*2048*128  bf16       =  4194304
    //   Vt  : 8*128*2048  bf16       =  4194304   (total ~12.75 MB)
    char*   ws = (char*)d_ws;
    __bf16* Wt = (__bf16*)ws;
    __bf16* Qb = (__bf16*)(ws + 786432);
    __bf16* Kb = Qb + 2097152;
    __bf16* Vt = Kb + 2097152;

    wprep_kernel<<<1536, 256, 0, stream>>>(Wq, Wk, Wv, Wt);
    qkv_proj_kernel<<<256, 256, 0, stream>>>(X, Wt, Qb, Kb, Vt);
    flash_attn_kernel<<<128, 256, 0, stream>>>(Qb, Kb, Vt, (float*)d_out);
}
